// CustomTransformerEncoderMoELayerStoich_62466004353101
// MI455X (gfx1250) — compile-verified, matrix operand order checked by reading, not run on hardware
//
#include <hip/hip_runtime.h>
#include <hip/hip_bf16.h>

// ---------------------------------------------------------------------------
// CDNA5 / gfx1250 implementation of the transformer-encoder + MoE layer.
// All large GEMMs run on v_wmma_f32_16x16x32_bf16 (f32 accumulate), 16x64
// output tile per wave (4 accumulators) for 4x A-fragment reuse.
// MoE input gather uses global_load_async_to_lds_b128 (ASYNCcnt path).
// ---------------------------------------------------------------------------

typedef __attribute__((ext_vector_type(16))) __bf16        bf16x16;
typedef __attribute__((ext_vector_type(8)))  float         f32x8;
typedef __attribute__((ext_vector_type(4)))  unsigned int  u32x4;

#define Bx   2
#define Tx   1024
#define Dx   768
#define FFx  3072
#define Ex   4
#define Hx   12
#define HDx  64
#define BTx  (Bx*Tx)

union Frag { bf16x16 v; u32x4 q[2]; };

// A/B fragment loader for 16-bit WMMA operands, row-major source with leading
// dim `ld` (elements). Per ISA: lane&15 selects the row (A: M, B: N); the two
// 16B chunks sit at k0+8*(lane>>4) and k0+16+8*(lane>>4).
__device__ __forceinline__ bf16x16 load_frag(const __bf16* p, int ld, int row0,
                                             int k0, int lane) {
  int m  = row0 + (lane & 15);
  int ks = (lane >> 4) << 3;
  const char* base = (const char*)(p + (size_t)m * ld + k0 + ks);
  Frag f;
  f.q[0] = *(const u32x4*)(base);
  f.q[1] = *(const u32x4*)(base + 32);   // +16 elements * 2B
  return f.v;
}

__device__ __forceinline__ f32x8 wmma_bf16(bf16x16 a, bf16x16 b, f32x8 c) {
  return __builtin_amdgcn_wmma_f32_16x16x32_bf16(false, a, false, b,
                                                 (short)0, c, false, false);
}

// ---------------------------------------------------------------------------
// Prep kernels
// ---------------------------------------------------------------------------
__global__ void cvt_kernel(const float* __restrict__ in, __bf16* __restrict__ out, int n) {
  int i = blockIdx.x * 256 + threadIdx.x;
  if (i < n) out[i] = (__bf16)in[i];
}

// W1 [E][D][FF] -> W1t [E][FF][D] bf16
__global__ void tw1_kernel(const float* __restrict__ W1, __bf16* __restrict__ W1t) {
  int i = blockIdx.x * 256 + threadIdx.x;
  if (i >= Ex * Dx * FFx) return;
  int e = i / (Dx * FFx); int rem = i % (Dx * FFx);
  int d = rem / FFx;      int f   = rem % FFx;
  W1t[((size_t)e * FFx + f) * Dx + d] = (__bf16)W1[i];
}

// W2 [E][FF][D] -> W2t [E][D][FF] bf16
__global__ void tw2_kernel(const float* __restrict__ W2, __bf16* __restrict__ W2t) {
  int i = blockIdx.x * 256 + threadIdx.x;
  if (i >= Ex * FFx * Dx) return;
  int e = i / (FFx * Dx); int rem = i % (FFx * Dx);
  int f = rem / Dx;       int d   = rem % Dx;
  W2t[((size_t)e * Dx + d) * FFx + f] = (__bf16)W2[i];
}

__global__ void zero_kernel(float* __restrict__ ff, int* __restrict__ counts) {
  size_t i = (size_t)blockIdx.x * 256 + threadIdx.x;
  if (i < (size_t)BTx * Dx) ff[i] = 0.f;
  if (i < Ex) counts[i] = 0;
}

// ---------------------------------------------------------------------------
// QKV projection: Y = X @ W^T + b.  One wave per 16x64 output tile
// (4 accumulators; A-fragment reused across the 4 N-subtiles).
// Q,K stored [b][h][t][64]; V stored transposed [b][h][64][t].
// ---------------------------------------------------------------------------
__global__ void qkv_kernel(const __bf16* __restrict__ X16,
                           const __bf16* __restrict__ Wq16,
                           const __bf16* __restrict__ Wk16,
                           const __bf16* __restrict__ Wv16,
                           const float* __restrict__ bq,
                           const float* __restrict__ bk,
                           const float* __restrict__ bv,
                           __bf16* __restrict__ Qh,
                           __bf16* __restrict__ Kh,
                           __bf16* __restrict__ Vt) {
  int tn = blockIdx.x, tm = blockIdx.y, z = blockIdx.z;   // tn: 64-col group
  int lane = threadIdx.x, hi = lane >> 4, nl = lane & 15;
  const __bf16* W    = (z == 0) ? Wq16 : (z == 1) ? Wk16 : Wv16;
  const float*  bias = (z == 0) ? bq   : (z == 1) ? bk   : bv;
  f32x8 acc[4] = {};
  for (int k0 = 0; k0 < Dx; k0 += 32) {
    bf16x16 a = load_frag(X16, Dx, tm * 16, k0, lane);
#pragma unroll
    for (int c = 0; c < 4; ++c) {
      bf16x16 b = load_frag(W, Dx, (tn * 4 + c) * 16, k0, lane);
      acc[c] = wmma_bf16(a, b, acc[c]);
    }
  }
#pragma unroll
  for (int c = 0; c < 4; ++c) {
    int   n  = (tn * 4 + c) * 16 + nl;
    float bs = bias[n];
    int   h  = n >> 6, j = n & 63;
#pragma unroll
    for (int r = 0; r < 8; ++r) {
      int   m = tm * 16 + r + hi * 8;
      int   b = m >> 10, t = m & 1023;
      float v = acc[c][r] + bs;
      if (z == 0)      Qh[(((size_t)b * Hx + h) * Tx + t) * HDx + j] = (__bf16)v;
      else if (z == 1) Kh[(((size_t)b * Hx + h) * Tx + t) * HDx + j] = (__bf16)v;
      else             Vt[(((size_t)b * Hx + h) * HDx + j) * Tx + t] = (__bf16)v;
    }
  }
}

// ---------------------------------------------------------------------------
// Attention: one wave per (b*h, 16-query-row block).
// logits (WMMA, Q-fragments hoisted) -> z-score -> softmax -> probs @ V^T
// (WMMA, 4 accumulators cover all 64 head-dim columns).
// Dynamic LDS: slog f32[16*1024] | sprob bf16[16*1024] | srow f32[16]
// ---------------------------------------------------------------------------
__global__ void attn_kernel(const __bf16* __restrict__ Qh,
                            const __bf16* __restrict__ Kh,
                            const __bf16* __restrict__ Vt,
                            const float* __restrict__ gptr,
                            __bf16* __restrict__ attn16) {
  extern __shared__ char smem[];
  float*  slog  = (float*)smem;                                 // 64 KB
  __bf16* sprob = (__bf16*)(smem + 16 * Tx * 4);                // 32 KB
  float*  srow  = (float*)(smem + 16 * Tx * 4 + 16 * Tx * 2);   // 64 B
  int bh = blockIdx.x, rowblk = blockIdx.y;
  int lane = threadIdx.x, hi = lane >> 4, nl = lane & 15;
  const __bf16* Qb = Qh + ((size_t)bh * Tx + rowblk * 16) * HDx;
  const __bf16* Kb = Kh + (size_t)bh * Tx * HDx;
  const float scale = 0.125f;   // hd^-0.5 = 64^-0.5

  // ---- logits into LDS; Q fragments are loop-invariant: hoist them ----
  bf16x16 qa0 = load_frag(Qb, HDx, 0, 0,  lane);
  bf16x16 qa1 = load_frag(Qb, HDx, 0, 32, lane);
  for (int nt = 0; nt < Tx / 16; ++nt) {
    f32x8 acc = {};
    bf16x16 b0 = load_frag(Kb, HDx, nt * 16, 0,  lane);
    bf16x16 b1 = load_frag(Kb, HDx, nt * 16, 32, lane);
    acc = wmma_bf16(qa0, b0, acc);
    acc = wmma_bf16(qa1, b1, acc);
#pragma unroll
    for (int r = 0; r < 8; ++r)
      slog[(r + hi * 8) * Tx + nt * 16 + nl] = acc[r] * scale;
  }
  __syncthreads();

  // ---- row stats: mean, unbiased std, min/max; each lane does half a row ----
  float gamma = gptr[0];
  int row = nl, half = hi;
  float* rp = slog + row * Tx + half * 512;
  float s = 0.f, ss = 0.f, mx = -3.4e38f, mn = 3.4e38f;
  for (int i = 0; i < 512; ++i) {
    float v = rp[i];
    s += v; ss += v * v; mx = fmaxf(mx, v); mn = fminf(mn, v);
  }
  s  += __shfl_xor(s, 16);  ss += __shfl_xor(ss, 16);
  mx  = fmaxf(mx, __shfl_xor(mx, 16));
  mn  = fminf(mn, __shfl_xor(mn, 16));
  float mean = s * (1.f / (float)Tx);
  float var  = (ss - (float)Tx * mean * mean) * (1.f / (float)(Tx - 1));
  float inv  = gamma / (sqrtf(var) + 1e-5f);      // z = (x-mean)*inv
  float zmax = fmaxf((mx - mean) * inv, (mn - mean) * inv);

  // ---- softmax numerator into bf16 prob buffer ----
  __bf16* pp = sprob + row * Tx + half * 512;
  float sumexp = 0.f;
  for (int i = 0; i < 512; ++i) {
    float z = (rp[i] - mean) * inv;
    float pv = __expf(z - zmax);
    sumexp += pv;
    pp[i] = (__bf16)pv;
  }
  sumexp += __shfl_xor(sumexp, 16);
  if (hi == 0) srow[row] = 1.f / sumexp;
  __syncthreads();

  // ---- attn = probs @ V; 4 accumulators = all 64 head-dim columns ----
  const __bf16* Vb = Vt + (size_t)bh * HDx * Tx;
  int b = bh / Hx, h = bh % Hx;
  f32x8 acc[4] = {};
  for (int k0 = 0; k0 < Tx; k0 += 32) {
    bf16x16 a = load_frag(sprob, Tx, 0, k0, lane);
#pragma unroll
    for (int c = 0; c < 4; ++c) {
      bf16x16 bfr = load_frag(Vb, Tx, c * 16, k0, lane);
      acc[c] = wmma_bf16(a, bfr, acc[c]);
    }
  }
#pragma unroll
  for (int c = 0; c < 4; ++c) {
    int j = c * 16 + nl;
#pragma unroll
    for (int r = 0; r < 8; ++r) {
      int m = r + hi * 8;
      int t = rowblk * 16 + m;
      float v = acc[c][r] * srow[m];
      attn16[((size_t)(b * Tx + t)) * Dx + h * HDx + j] = (__bf16)v;
    }
  }
}

// ---------------------------------------------------------------------------
// Output projection + residual: y1 = src + attn @ Wo^T + bo   (fp32 out)
// One wave per 16x64 tile.
// ---------------------------------------------------------------------------
__global__ void wo_kernel(const __bf16* __restrict__ attn16,
                          const __bf16* __restrict__ Wo16,
                          const float* __restrict__ bo,
                          const float* __restrict__ src,
                          float* __restrict__ y1) {
  int tn = blockIdx.x, tm = blockIdx.y;
  int lane = threadIdx.x, hi = lane >> 4, nl = lane & 15;
  f32x8 acc[4] = {};
  for (int k0 = 0; k0 < Dx; k0 += 32) {
    bf16x16 a = load_frag(attn16, Dx, tm * 16, k0, lane);
#pragma unroll
    for (int c = 0; c < 4; ++c) {
      bf16x16 b = load_frag(Wo16, Dx, (tn * 4 + c) * 16, k0, lane);
      acc[c] = wmma_bf16(a, b, acc[c]);
    }
  }
#pragma unroll
  for (int c = 0; c < 4; ++c) {
    int n = (tn * 4 + c) * 16 + nl;
    float bias = bo[n];
#pragma unroll
    for (int r = 0; r < 8; ++r) {
      size_t m = tm * 16 + r + hi * 8;
      y1[m * Dx + n] = acc[c][r] + bias + src[m * Dx + n];
    }
  }
}

// ---------------------------------------------------------------------------
// LayerNorm over D=768 (optional residual in = a + b, optional bf16 copy out)
// ---------------------------------------------------------------------------
__global__ void ln_kernel(const float* __restrict__ a, const float* __restrict__ b,
                          const float* __restrict__ g, const float* __restrict__ beta,
                          float* __restrict__ outf, __bf16* __restrict__ outb) {
  __shared__ float r1[256], r2[256];
  int row = blockIdx.x, tid = threadIdx.x;
  const float* pa = a + (size_t)row * Dx;
  const float* pb = b ? b + (size_t)row * Dx : nullptr;
  float vals[3];
  float s = 0.f, ss = 0.f;
  for (int i = 0; i < 3; ++i) {
    int c = tid + i * 256;
    float v = pa[c] + (pb ? pb[c] : 0.f);
    vals[i] = v; s += v; ss += v * v;
  }
  r1[tid] = s; r2[tid] = ss;
  __syncthreads();
  for (int off = 128; off; off >>= 1) {
    if (tid < off) { r1[tid] += r1[tid + off]; r2[tid] += r2[tid + off]; }
    __syncthreads();
  }
  float mean = r1[0] * (1.f / (float)Dx);
  float var  = r2[0] * (1.f / (float)Dx) - mean * mean;
  float rstd = rsqrtf(var + 1e-5f);
  for (int i = 0; i < 3; ++i) {
    int c = tid + i * 256;
    float o = (vals[i] - mean) * rstd * g[c] + beta[c];
    if (outf) outf[(size_t)row * Dx + c] = o;
    if (outb) outb[(size_t)row * Dx + c] = (__bf16)o;
  }
}

// ---------------------------------------------------------------------------
// MoE gate: softmax over E=4, top-2, build per-expert token lists.
// ---------------------------------------------------------------------------
__global__ void gate_kernel(const float* __restrict__ x,
                            const float* __restrict__ Wg,
                            const float* __restrict__ bg,
                            int* __restrict__ counts,
                            int* __restrict__ listTok,
                            float* __restrict__ listW) {
  int t = blockIdx.x, lane = threadIdx.x;
  const float* xr = x + (size_t)t * Dx;
  float a0 = 0.f, a1 = 0.f, a2 = 0.f, a3 = 0.f;
  for (int c = lane; c < Dx; c += 32) {
    float xv = xr[c];
    a0 += xv * Wg[c];
    a1 += xv * Wg[Dx + c];
    a2 += xv * Wg[2 * Dx + c];
    a3 += xv * Wg[3 * Dx + c];
  }
  for (int off = 16; off; off >>= 1) {
    a0 += __shfl_xor(a0, off); a1 += __shfl_xor(a1, off);
    a2 += __shfl_xor(a2, off); a3 += __shfl_xor(a3, off);
  }
  if (lane == 0) {
    float l[Ex] = {a0 + bg[0], a1 + bg[1], a2 + bg[2], a3 + bg[3]};
    float mx = fmaxf(fmaxf(l[0], l[1]), fmaxf(l[2], l[3]));
    float e[Ex], sum = 0.f;
    for (int i = 0; i < Ex; ++i) { e[i] = __expf(l[i] - mx); sum += e[i]; }
    float invs = 1.f / sum;
    int i0 = 0;
    for (int i = 1; i < Ex; ++i) if (e[i] > e[i0]) i0 = i;
    int i1 = -1;
    for (int i = 0; i < Ex; ++i) {
      if (i == i0) continue;
      if (i1 < 0 || e[i] > e[i1]) i1 = i;
    }
    int s0 = atomicAdd(&counts[i0], 1);
    listTok[i0 * 4096 + s0] = t; listW[i0 * 4096 + s0] = e[i0] * invs;
    int s1 = atomicAdd(&counts[i1], 1);
    listTok[i1 * 4096 + s1] = t; listW[i1 * 4096 + s1] = e[i1] * invs;
  }
}

// ---------------------------------------------------------------------------
// MoE FFN: per (expert, 16-token tile).  4 waves / block.
// Token-row gather uses CDNA5 async global->LDS DMA (ASYNCcnt), then both
// GEMMs run 16x64-per-wave WMMA tiles.
// LDS: sx bf16[16*768] | sh bf16[16*3072] | stok i32[16] | sw f32[16]
// ---------------------------------------------------------------------------
__global__ void moe_kernel(const __bf16* __restrict__ x16,
                           const __bf16* __restrict__ W1t,
                           const __bf16* __restrict__ W2t,
                           const float* __restrict__ b1,
                           const float* __restrict__ b2,
                           const int* __restrict__ counts,
                           const int* __restrict__ listTok,
                           const float* __restrict__ listW,
                           float* __restrict__ ff) {
  extern __shared__ char smem[];
  __bf16* sx   = (__bf16*)smem;                                  // 24 KB
  __bf16* sh   = (__bf16*)(smem + 16 * Dx * 2);                  // 96 KB
  int*    stok = (int*)(smem + 16 * Dx * 2 + 16 * FFx * 2);
  float*  sw   = (float*)(stok + 16);

  int e = blockIdx.y, tx = blockIdx.x;
  int count = counts[e];
  if (tx * 16 >= count) return;                 // uniform across block
  int tid = threadIdx.x;

  if (tid < 16) {
    int slot = tx * 16 + tid;
    int tok = 0; float w = 0.f;
    if (slot < count) { tok = listTok[e * 4096 + slot]; w = listW[e * 4096 + slot]; }
    stok[tid] = tok; sw[tid] = w;
  }
  __syncthreads();

  // ---- gather 16 token rows (192B/thread) via async global->LDS B128 DMA.
  // ISA 08_async_tensor §4.4: the instruction offset is added to BOTH the LDS
  // destination (VGPR lds byte-offset) and the global source address, so one
  // address pair + immediate offsets covers the whole contiguous copy.
  {
    int r = tid >> 3, c0 = (tid & 7) * 96;      // 96 elems = 192B per thread
    unsigned       ldsoff = (unsigned)((char*)(sx + r * Dx + c0) - smem);
    unsigned long long ga = (unsigned long long)(x16 + (size_t)stok[r] * Dx + c0);
#pragma unroll
    for (int i = 0; i < 12; ++i) {
      asm volatile("global_load_async_to_lds_b128 %0, %1, off offset:%2"
                   :: "v"(ldsoff), "v"(ga), "i"(i * 16) : "memory");
    }
    asm volatile("s_wait_asynccnt 0x0" ::: "memory");
  }
  __syncthreads();

  int wid = tid >> 5, lane = tid & 31;
  int hi = lane >> 4, nl = lane & 15;

  // GEMM1: h = relu(x @ W1[e] + b1[e])   (M=16, N=3072, K=768)
  // 192 N-tiles in groups of 4 (16x64 per wave), strided across 4 waves.
  const __bf16* W1e = W1t + (size_t)e * FFx * Dx;
  for (int g = wid; g < FFx / 64; g += 4) {
    f32x8 acc[4] = {};
    for (int k0 = 0; k0 < Dx; k0 += 32) {
      bf16x16 a = load_frag(sx, Dx, 0, k0, lane);
#pragma unroll
      for (int c = 0; c < 4; ++c) {
        bf16x16 b = load_frag(W1e, Dx, (g * 4 + c) * 16, k0, lane);
        acc[c] = wmma_bf16(a, b, acc[c]);
      }
    }
#pragma unroll
    for (int c = 0; c < 4; ++c) {
      int n = (g * 4 + c) * 16 + nl;
      float bias = b1[e * FFx + n];
#pragma unroll
      for (int r = 0; r < 8; ++r) {
        float v = fmaxf(acc[c][r] + bias, 0.f);
        sh[(r + hi * 8) * FFx + n] = (__bf16)v;
      }
    }
  }
  __syncthreads();

  // GEMM2: y = h @ W2[e] + b2[e]; scatter w*y into ff   (M=16, N=768, K=3072)
  const __bf16* W2e = W2t + (size_t)e * Dx * FFx;
  for (int g = wid; g < Dx / 64; g += 4) {
    f32x8 acc[4] = {};
    for (int k0 = 0; k0 < FFx; k0 += 32) {
      bf16x16 a = load_frag(sh, FFx, 0, k0, lane);
#pragma unroll
      for (int c = 0; c < 4; ++c) {
        bf16x16 b = load_frag(W2e, FFx, (g * 4 + c) * 16, k0, lane);
        acc[c] = wmma_bf16(a, b, acc[c]);
      }
    }
#pragma unroll
    for (int c = 0; c < 4; ++c) {
      int n = (g * 4 + c) * 16 + nl;
      float bias = b2[e * Dx + n];
#pragma unroll
      for (int r = 0; r < 8; ++r) {
        int m = r + hi * 8;
        float w = sw[m];
        if (w != 0.f)
          atomicAdd(&ff[(size_t)stok[m] * Dx + n], w * (acc[c][r] + bias));
      }
    }
  }
}

// ---------------------------------------------------------------------------
// Host-side orchestration (all launches on `stream`; graph-capture safe)
// ---------------------------------------------------------------------------
extern "C" void kernel_launch(void* const* d_in, const int* in_sizes, int n_in,
                              void* d_out, int out_size, void* d_ws, size_t ws_size,
                              hipStream_t stream) {
  const float* src  = (const float*)d_in[0];
  // d_in[1] = frac: unused (reference's add_frac_bias=False path)
  const float* Wq   = (const float*)d_in[2];  const float* bq   = (const float*)d_in[3];
  const float* Wk   = (const float*)d_in[4];  const float* bk   = (const float*)d_in[5];
  const float* Wv   = (const float*)d_in[6];  const float* bv   = (const float*)d_in[7];
  const float* Wo   = (const float*)d_in[8];  const float* bo   = (const float*)d_in[9];
  const float* gam  = (const float*)d_in[10];
  const float* ln1g = (const float*)d_in[11]; const float* ln1b = (const float*)d_in[12];
  const float* ln2g = (const float*)d_in[13]; const float* ln2b = (const float*)d_in[14];
  const float* Wg   = (const float*)d_in[15]; const float* bg   = (const float*)d_in[16];
  const float* W1   = (const float*)d_in[17]; const float* b1   = (const float*)d_in[18];
  const float* W2   = (const float*)d_in[19]; const float* b2   = (const float*)d_in[20];

  char* p = (char*)d_ws;
  auto alloc = [&](size_t bytes) -> void* {
    void* r = (void*)p;
    p += (bytes + 255) & ~(size_t)255;
    return r;
  };
  __bf16* X16     = (__bf16*)alloc((size_t)BTx * Dx * 2);
  __bf16* Wq16    = (__bf16*)alloc((size_t)Dx * Dx * 2);
  __bf16* Wk16    = (__bf16*)alloc((size_t)Dx * Dx * 2);
  __bf16* Wv16    = (__bf16*)alloc((size_t)Dx * Dx * 2);
  __bf16* Wo16    = (__bf16*)alloc((size_t)Dx * Dx * 2);
  __bf16* W1t     = (__bf16*)alloc((size_t)Ex * FFx * Dx * 2);
  __bf16* W2t     = (__bf16*)alloc((size_t)Ex * Dx * FFx * 2);
  __bf16* Qh      = (__bf16*)alloc((size_t)Bx * Hx * Tx * HDx * 2);
  __bf16* Kh      = (__bf16*)alloc((size_t)Bx * Hx * Tx * HDx * 2);
  __bf16* Vt      = (__bf16*)alloc((size_t)Bx * Hx * HDx * Tx * 2);
  __bf16* attn16  = (__bf16*)alloc((size_t)BTx * Dx * 2);
  float*  y1      = (float*) alloc((size_t)BTx * Dx * 4);
  float*  xf      = (float*) alloc((size_t)BTx * Dx * 4);
  __bf16* x16     = (__bf16*)alloc((size_t)BTx * Dx * 2);
  float*  ffbuf   = (float*) alloc((size_t)BTx * Dx * 4);
  int*    counts  = (int*)   alloc(256);
  int*    listTok = (int*)   alloc((size_t)Ex * 4096 * 4);
  float*  listW   = (float*) alloc((size_t)Ex * 4096 * 4);
  (void)ws_size; (void)n_in; (void)in_sizes; (void)out_size;

  int n;
  // ---- precision / layout prep ----
  n = BTx * Dx;     cvt_kernel<<<(n + 255) / 256, 256, 0, stream>>>(src, X16, n);
  n = Dx * Dx;      cvt_kernel<<<(n + 255) / 256, 256, 0, stream>>>(Wq, Wq16, n);
                    cvt_kernel<<<(n + 255) / 256, 256, 0, stream>>>(Wk, Wk16, n);
                    cvt_kernel<<<(n + 255) / 256, 256, 0, stream>>>(Wv, Wv16, n);
                    cvt_kernel<<<(n + 255) / 256, 256, 0, stream>>>(Wo, Wo16, n);
  n = Ex * Dx * FFx; tw1_kernel<<<(n + 255) / 256, 256, 0, stream>>>(W1, W1t);
  n = Ex * FFx * Dx; tw2_kernel<<<(n + 255) / 256, 256, 0, stream>>>(W2, W2t);

  // ---- attention block ----
  qkv_kernel<<<dim3(Dx / 64, BTx / 16, 3), 32, 0, stream>>>(
      X16, Wq16, Wk16, Wv16, bq, bk, bv, Qh, Kh, Vt);

  size_t attn_lds = (size_t)16 * Tx * 4 + (size_t)16 * Tx * 2 + 64;
  attn_kernel<<<dim3(Bx * Hx, Tx / 16), 32, attn_lds, stream>>>(
      Qh, Kh, Vt, gam, attn16);

  wo_kernel<<<dim3(Dx / 64, BTx / 16), 32, 0, stream>>>(attn16, Wo16, bo, src, y1);

  // ---- residual + LN1 (also bf16 copy for MoE GEMMs) ----
  ln_kernel<<<BTx, 256, 0, stream>>>(y1, nullptr, ln1g, ln1b, xf, x16);

  // ---- MoE top-2 ----
  zero_kernel<<<(BTx * Dx + 255) / 256, 256, 0, stream>>>(ffbuf, counts);
  gate_kernel<<<BTx, 32, 0, stream>>>(xf, Wg, bg, counts, listTok, listW);

  size_t moe_lds = (size_t)16 * Dx * 2 + (size_t)16 * FFx * 2 + 16 * 4 + 16 * 4;
  moe_kernel<<<dim3(256, Ex), 128, moe_lds, stream>>>(
      x16, W1t, W2t, b1, b2, counts, listTok, listW, ffbuf);

  // ---- residual + LN2 -> output (fp32) ----
  ln_kernel<<<BTx, 256, 0, stream>>>(xf, ffbuf, ln2g, ln2b, (float*)d_out, nullptr);
}
